// BiLstm_48550310314336
// MI455X (gfx1250) — compile-verified, matrix operand order checked by reading, not run on hardware
//
#include <hip/hip_runtime.h>
#include <hip/hip_bf16.h>
#include <math.h>
#include <stdint.h>

typedef __bf16 bf16;
typedef __attribute__((ext_vector_type(16))) __bf16 v16bf;
typedef __attribute__((ext_vector_type(8)))  __bf16 v8bf;
typedef __attribute__((ext_vector_type(8)))  float  v8f;
typedef __attribute__((ext_vector_type(4)))  unsigned int u32x4;
typedef __attribute__((ext_vector_type(8)))  int i32x8;
typedef __attribute__((ext_vector_type(4)))  int i32x4;

#define Tt   512
#define Bb   32
#define Vv   128
#define Hh   512
#define G4H  2048
#define ATTN 20
#define EPSV 1e-5f

__device__ __forceinline__ float sigmoidf_(float x){ return 1.f/(1.f+__expf(-x)); }

// ---------------- BN1: per-channel stats over (B,T) ----------------
__global__ void bn1_stats(const float* __restrict__ x, float* __restrict__ mu,
                          float* __restrict__ istd){
  int v = blockIdx.x, tid = threadIdx.x;
  float s = 0.f, s2 = 0.f;
  for (int i = tid; i < Bb*Tt; i += 256){
    int b = i / Tt, t = i % Tt;
    float val = x[((size_t)b*Tt + t)*Vv + v];
    s += val; s2 += val*val;
  }
  __shared__ float rs[256], rq[256];
  rs[tid] = s; rq[tid] = s2; __syncthreads();
  for (int o = 128; o > 0; o >>= 1){
    if (tid < o){ rs[tid] += rs[tid+o]; rq[tid] += rq[tid+o]; }
    __syncthreads();
  }
  if (tid == 0){
    float m = rs[0] / (float)(Bb*Tt);
    float var = rq[0] / (float)(Bb*Tt) - m*m;
    mu[v] = m; istd[v] = rsqrtf(var + EPSV);
  }
}

// normalize x and re-layout time-major [t][b][v] as bf16
__global__ void bn1_apply(const float* __restrict__ x, const float* __restrict__ mu,
                          const float* __restrict__ istd, const float* __restrict__ g,
                          const float* __restrict__ be, bf16* __restrict__ bnx){
  int idx = blockIdx.x*blockDim.x + threadIdx.x;
  if (idx >= Tt*Bb*Vv) return;
  int t = idx / (Bb*Vv);
  int r = idx % (Bb*Vv);
  int b = r / Vv, v = r % Vv;
  float val = x[((size_t)b*Tt + t)*Vv + v];
  val = (val - mu[v]) * istd[v] * g[v] + be[v];
  bnx[idx] = (bf16)val;   // bnx[(t*32+b)*128+v]
}

// ---------------- Weight packing: fp32 -> bf16 WMMA B-fragment layout ---------
__global__ void pack_w(const float* __restrict__ w_hh, const float* __restrict__ w_ih,
                       bf16* __restrict__ out, int Kc, int Vin, int total){
  int idx = blockIdx.x*blockDim.x + threadIdx.x;
  if (idx >= total) return;
  int Ktiles = Kc / 32;
  int e    = idx & 511;
  int tile = (idx >> 9) % (Ktiles*128);
  int d    = idx / (512*Ktiles*128);
  int nt = tile / Ktiles, kt = tile % Ktiles;
  int l = e >> 4, s = e & 15, vv = s >> 1, hf = s & 1;
  int l15 = l & 15, hi = l >> 4;
  int kl = (vv < 4) ? (2*vv + hf + 8*hi) : (16 + 2*(vv-4) + hf + 8*hi);
  int k = kt*32 + kl, n = nt*16 + l15;
  float val;
  if (k < Hh) val = w_hh[((size_t)d*G4H + n)*Hh  + k];
  else        val = w_ih[((size_t)d*G4H + n)*Vin + (k - Hh)];
  out[idx] = (bf16)val;
}

// ---------------- Recurrent bi-LSTM layer: one workgroup per direction -------
// Templated so B-tile addressing folds into 24-bit immediates; x_t staged via
// the Tensor Data Mover (TDM) when available; all state + LN params in LDS.
// B fragments double-buffered (depth 2) to stay under the no-spill VGPR budget
// at 8 waves/SIMD.
template<int KC, int VIN>
__global__ __launch_bounds__(1024)
void lstm_layer(const bf16* __restrict__ Wpack,  // [2][128 nt][KTILES][512]
                const bf16* __restrict__ xin,    // [T][B][VIN] bf16
                const float* __restrict__ gg_, const float* __restrict__ gb_, // [2][4H]
                const float* __restrict__ cg_, const float* __restrict__ cb_, // [2][H]
                bf16* __restrict__ hout){        // [T][B][2H], write cols dir*H..
  constexpr int KTILES = KC / 32;
  const int dir = blockIdx.x;
  extern __shared__ char smem[];
  bf16*  hbuf = (bf16*)smem;             // [32][512]  hidden state (A cols 0..511)
  bf16*  xbuf = hbuf + Bb*Hh;            // [32][VIN]  x_t          (A cols 512..)
  bf16*  act  = xbuf + Bb*VIN;           // [32][2048] gate activations
  float* cbuf = (float*)(act + Bb*G4H);  // [32][512]  cell state
  float* ggs  = cbuf + Bb*Hh;            // [2048] gate LN gamma
  float* gbs  = ggs + G4H;               // [2048] gate LN beta
  float* cgs  = gbs + G4H;               // [512]  cell LN gamma
  float* cbs  = cgs + Hh;                // [512]  cell LN beta
  float* gstat = cbs + Hh;               // [4][32][2] gate-LN sums
  float* cstat = gstat + 4*Bb*2;         // [32][2]    cell-LN sums

  const bf16* Wd = Wpack + (size_t)dir * KTILES * 128 * 512;
  const int tid = threadIdx.x, wave = tid >> 5, lane = tid & 31;
  const int l15 = lane & 15, hi = lane >> 4;
  const int mt = wave & 1;            // M tile (batch rows 16*mt..)
  const int nb = (wave >> 1) * 8;     // first of 8 N-tiles (128 contiguous cols)
  const int gate_w = wave >> 3;       // wave's 128-col block lies in one gate

  // init state + cache LN params in LDS
  for (int i = tid; i < Bb*Hh; i += 1024){ hbuf[i] = (bf16)0.f; cbuf[i] = 0.f; }
  for (int i = tid; i < G4H; i += 1024){ ggs[i] = gg_[dir*G4H + i]; gbs[i] = gb_[dir*G4H + i]; }
  for (int i = tid; i < Hh;  i += 1024){ cgs[i] = cg_[dir*Hh + i]; cbs[i] = cb_[dir*Hh + i]; }
  __syncthreads();

  for (int t = 0; t < Tt; ++t){
    const int tt = dir ? (Tt - 1 - t) : t;
    const bf16* xrow = xin + (size_t)tt * Bb * VIN;

    // ---- stage x_t into xbuf via Tensor Data Mover (2D tile 32 x VIN, 2B) ----
#if __has_builtin(__builtin_amdgcn_tensor_load_to_lds)
    if (wave == 0){
      unsigned long long ga = (unsigned long long)(uintptr_t)xrow;
      unsigned ldsa = (unsigned)(uintptr_t)xbuf;
      u32x4 g0 = { 1u,                                   // count=1, user descriptor
                   ldsa,                                 // lds_addr
                   (unsigned)(ga & 0xFFFFFFFFull),       // global_addr[31:0]
                   (unsigned)((ga >> 32) & 0x01FFFFFFull) | (2u << 30) }; // addr hi + type=2
      i32x8 g1 = { (int)(1u << 16),                      // data_size = 2 bytes
                   (int)((unsigned)VIN << 16),           // tensor_dim0 (low16 @ bit48)
                   (int)(32u << 16),                     // tensor_dim1 = 32 rows
                   (int)((unsigned)VIN << 16),           // tile_dim0 = VIN
                   32,                                   // tile_dim1 = 32
                   VIN,                                  // tensor_dim0_stride (48b, low32)
                   0, 0 };
      i32x4 gz = {0,0,0,0};
#if defined(__clang_major__) && (__clang_major__ >= 23)
      i32x8 gz8 = {0,0,0,0,0,0,0,0};
      __builtin_amdgcn_tensor_load_to_lds(g0, g1, gz, gz, gz8, 0);
#else
      __builtin_amdgcn_tensor_load_to_lds(g0, g1, gz, gz, 0);
#endif
      __builtin_amdgcn_s_wait_tensorcnt(0);
    }
#else
    for (int i = tid; i < Bb*VIN; i += 1024) xbuf[i] = xrow[i];
#endif
    if (tid < 4*Bb*2) gstat[tid] = 0.f;
    if (tid < Bb*2)   cstat[tid] = 0.f;
    __syncthreads();

    // ---- pre = [h|x] @ W^T  (32 x 2048), 8 C-tiles per wave ----
    v8f acc[8];
    #pragma unroll
    for (int i = 0; i < 8; ++i)
      #pragma unroll
      for (int r = 0; r < 8; ++r) acc[i][r] = 0.f;

    const int arow = mt*16 + l15;
    union BU { v16bf v; v8bf h[2]; };

    // one kt slice: A fragment from LDS, 8 B tiles at immediate offsets,
    // depth-2 pipeline (load tile i+1 while WMMA on tile i)
    auto kt_slice = [&](const bf16* ap, int kt){
      BU au;
      au.h[0] = *(const v8bf*)(ap);
      au.h[1] = *(const v8bf*)(ap + 16);
      const bf16* wpb = Wd + ((size_t)nb*KTILES + kt)*512 + lane*16;
      BU bu[2];
      __builtin_prefetch(wpb + 512, 0, 3);               // next kt tile (contiguous)
      bu[0].h[0] = *(const v8bf*)(wpb);
      bu[0].h[1] = *(const v8bf*)(wpb + 8);
      #pragma unroll
      for (int i = 0; i < 8; ++i){
        if (i < 7){
          const bf16* wp = wpb + (size_t)(i+1)*(KTILES*512);
          __builtin_prefetch(wp + 512, 0, 3);
          bu[(i+1)&1].h[0] = *(const v8bf*)(wp);
          bu[(i+1)&1].h[1] = *(const v8bf*)(wp + 8);
        }
        acc[i] = __builtin_amdgcn_wmma_f32_16x16x32_bf16(
                     false, au.v, false, bu[i&1].v, (short)0, acc[i], false, false);
      }
    };
    for (int kt = 0; kt < 16; ++kt)
      kt_slice(hbuf + arow*Hh + kt*32 + hi*8, kt);
    for (int kt = 16; kt < KTILES; ++kt)
      kt_slice(xbuf + arow*VIN + (kt-16)*32 + hi*8, kt);

    // ---- gate LayerNorm stats: sum/sumsq per (gate, batch) over H cols ----
    #pragma unroll
    for (int r = 0; r < 8; ++r){
      float s = 0.f, s2 = 0.f;
      #pragma unroll
      for (int i = 0; i < 8; ++i){ float v = acc[i][r]; s += v; s2 += v*v; }
      int b = mt*16 + r + 8*hi;
      atomicAdd(&gstat[(gate_w*Bb + b)*2 + 0], s);
      atomicAdd(&gstat[(gate_w*Bb + b)*2 + 1], s2);
    }
    __syncthreads();

    // ---- normalize + gate nonlinearity -> act (bf16) ----
    #pragma unroll
    for (int r = 0; r < 8; ++r){
      int b = mt*16 + r + 8*hi;
      float m   = gstat[(gate_w*Bb + b)*2 + 0] * (1.f/(float)Hh);
      float var = gstat[(gate_w*Bb + b)*2 + 1] * (1.f/(float)Hh) - m*m;
      float is  = rsqrtf(var + EPSV);
      #pragma unroll
      for (int i = 0; i < 8; ++i){
        int colg = (wave >> 1)*128 + i*16 + l15;
        float vn = (acc[i][r] - m) * is * ggs[colg] + gbs[colg];
        float a  = (gate_w == 2) ? tanhf(vn) : sigmoidf_(vn);
        act[b*G4H + colg] = (bf16)a;
      }
    }
    __syncthreads();

    // ---- c = sig(f)*c + sig(i)*tanh(g); cell-LN stats ----
    #pragma unroll
    for (int k2 = 0; k2 < 16; ++k2){
      int idx = tid + k2*1024;
      int b = idx >> 9, j = idx & 511;
      float iv = (float)act[b*G4H + j];
      float fv = (float)act[b*G4H + 512  + j];
      float gv = (float)act[b*G4H + 1024 + j];
      float c  = fv * cbuf[b*Hh + j] + iv * gv;
      cbuf[b*Hh + j] = c;
      atomicAdd(&cstat[b*2 + 0], c);
      atomicAdd(&cstat[b*2 + 1], c*c);
    }
    __syncthreads();

    // ---- h = sig(o)*tanh(LN(c)); feed back + emit ----
    #pragma unroll
    for (int k2 = 0; k2 < 16; ++k2){
      int idx = tid + k2*1024;
      int b = idx >> 9, j = idx & 511;
      float c   = cbuf[b*Hh + j];
      float m   = cstat[b*2 + 0] * (1.f/(float)Hh);
      float var = cstat[b*2 + 1] * (1.f/(float)Hh) - m*m;
      float cn  = (c - m) * rsqrtf(var + EPSV) * cgs[j] + cbs[j];
      float ov  = (float)act[b*G4H + 1536 + j];
      float h   = ov * tanhf(cn);
      hbuf[b*Hh + j] = (bf16)h;
      hout[((size_t)tt*Bb + b)*(2*Hh) + dir*Hh + j] = (bf16)h;
    }
    __syncthreads();
  }
}

// ---------------- BN2 stats over layer-1 output --------------------
__global__ void bn2_stats(const bf16* __restrict__ h1, float* __restrict__ mu,
                          float* __restrict__ istd){
  int c = blockIdx.x, tid = threadIdx.x;
  float s = 0.f, s2 = 0.f;
  for (int i = tid; i < Tt*Bb; i += 256){
    float v = (float)h1[(size_t)i*(2*Hh) + c];
    s += v; s2 += v*v;
  }
  __shared__ float rs[256], rq[256];
  rs[tid] = s; rq[tid] = s2; __syncthreads();
  for (int o = 128; o > 0; o >>= 1){
    if (tid < o){ rs[tid] += rs[tid+o]; rq[tid] += rq[tid+o]; }
    __syncthreads();
  }
  if (tid == 0){
    float m = rs[0] / (float)(Tt*Bb);
    float var = rq[0] / (float)(Tt*Bb) - m*m;
    mu[c] = m; istd[c] = rsqrtf(var + EPSV);
  }
}

// ---------------- Attention + weighted sum: one block per batch ----
__global__ __launch_bounds__(256)
void attn_out(const bf16* __restrict__ h1, const float* __restrict__ mu,
              const float* __restrict__ istd, const float* __restrict__ g,
              const float* __restrict__ be, const float* __restrict__ w_om,
              const float* __restrict__ b_om, const float* __restrict__ u_om,
              float* __restrict__ out){
  int b = blockIdx.x, tid = threadIdx.x;
  __shared__ float vu[Tt];
  __shared__ float red[256];

  for (int t = tid; t < Tt; t += 256){
    float acc[ATTN];
    #pragma unroll
    for (int a = 0; a < ATTN; ++a) acc[a] = b_om[a];
    const bf16* hrow = h1 + ((size_t)t*Bb + b)*(2*Hh);
    for (int c = 0; c < 2*Hh; ++c){
      float hn = ((float)hrow[c] - mu[c]) * istd[c] * g[c] + be[c];
      #pragma unroll
      for (int a = 0; a < ATTN; ++a) acc[a] += hn * w_om[c*ATTN + a];
    }
    float s = 0.f;
    #pragma unroll
    for (int a = 0; a < ATTN; ++a) s += tanhf(acc[a]) * u_om[a];
    vu[t] = s;
  }
  __syncthreads();

  float mx = -1e30f;
  for (int t = tid; t < Tt; t += 256) mx = fmaxf(mx, vu[t]);
  red[tid] = mx; __syncthreads();
  for (int o = 128; o > 0; o >>= 1){
    if (tid < o) red[tid] = fmaxf(red[tid], red[tid+o]);
    __syncthreads();
  }
  mx = red[0]; __syncthreads();
  float sm = 0.f;
  for (int t = tid; t < Tt; t += 256){ float e = __expf(vu[t]-mx); vu[t] = e; sm += e; }
  red[tid] = sm; __syncthreads();
  for (int o = 128; o > 0; o >>= 1){
    if (tid < o) red[tid] += red[tid+o];
    __syncthreads();
  }
  float inv = 1.f / red[0];

  for (int c = tid; c < 2*Hh; c += 256){
    float acc = 0.f;
    for (int t = 0; t < Tt; ++t){
      float hn = ((float)h1[((size_t)t*Bb + b)*(2*Hh) + c] - mu[c]) * istd[c] * g[c] + be[c];
      acc += vu[t] * hn;
    }
    out[(size_t)b*(2*Hh) + c] = acc * inv;
  }
}

// ---------------- launch ----------------
extern "C" void kernel_launch(void* const* d_in, const int* in_sizes, int n_in,
                              void* d_out, int out_size, void* d_ws, size_t ws_size,
                              hipStream_t stream){
  (void)in_sizes; (void)n_in; (void)out_size; (void)ws_size;
  const float* x      = (const float*)d_in[0];
  const float* bn1_g  = (const float*)d_in[1];
  const float* bn1_b  = (const float*)d_in[2];
  const float* bn2_g  = (const float*)d_in[3];
  const float* bn2_b  = (const float*)d_in[4];
  const float* w_ih0  = (const float*)d_in[5];
  const float* w_hh0  = (const float*)d_in[6];
  const float* g_g0   = (const float*)d_in[7];
  const float* g_b0   = (const float*)d_in[8];
  const float* c_g0   = (const float*)d_in[9];
  const float* c_b0   = (const float*)d_in[10];
  const float* w_ih1  = (const float*)d_in[11];
  const float* w_hh1  = (const float*)d_in[12];
  const float* g_g1   = (const float*)d_in[13];
  const float* g_b1   = (const float*)d_in[14];
  const float* c_g1   = (const float*)d_in[15];
  const float* c_b1   = (const float*)d_in[16];
  const float* w_om   = (const float*)d_in[17];
  const float* b_om   = (const float*)d_in[18];
  const float* u_om   = (const float*)d_in[19];
  float* out = (float*)d_out;

  char* ws = (char*)d_ws;
  size_t off = 0;
  auto take = [&](size_t bytes)->char*{
    char* p = ws + off; off += (bytes + 255) & ~(size_t)255; return p;
  };
  float* mu1   = (float*)take(Vv*4);
  float* istd1 = (float*)take(Vv*4);
  float* mu2   = (float*)take(2*Hh*4);
  float* istd2 = (float*)take(2*Hh*4);
  bf16* bnx = (bf16*)take((size_t)Tt*Bb*Vv*2);
  const int Kc0 = Hh + Vv;        // 640,  20 K-tiles
  const int Kc1 = Hh + 2*Hh;      // 1536, 48 K-tiles
  bf16* Wp0 = (bf16*)take((size_t)2*(Kc0/32)*128*512*2);
  bf16* Wp1 = (bf16*)take((size_t)2*(Kc1/32)*128*512*2);
  bf16* h0  = (bf16*)take((size_t)Tt*Bb*2*Hh*2);
  bf16* h1  = (bf16*)take((size_t)Tt*Bb*2*Hh*2);

  // BN1
  bn1_stats<<<Vv, 256, 0, stream>>>(x, mu1, istd1);
  {
    int tot = Tt*Bb*Vv;
    bn1_apply<<<(tot+255)/256, 256, 0, stream>>>(x, mu1, istd1, bn1_g, bn1_b, bnx);
  }
  // pack weights (per layer, both dirs)
  {
    int tot0 = 2*(Kc0/32)*128*512;
    pack_w<<<(tot0+255)/256, 256, 0, stream>>>(w_hh0, w_ih0, Wp0, Kc0, Vv, tot0);
    int tot1 = 2*(Kc1/32)*128*512;
    pack_w<<<(tot1+255)/256, 256, 0, stream>>>(w_hh1, w_ih1, Wp1, Kc1, 2*Hh, tot1);
  }
  // recurrent layers (grid = 2 directions)
  auto shbytes = [](int VIN)->size_t{
    return (size_t)Bb*Hh*2 + (size_t)Bb*VIN*2 + (size_t)Bb*G4H*2 + (size_t)Bb*Hh*4
         + (size_t)(G4H*2 + Hh*2 + 4*Bb*2 + Bb*2)*4;
  };
  size_t sh0 = shbytes(Vv);     // 259,328 B
  size_t sh1 = shbytes(2*Hh);   // 316,672 B  (needs CDNA5 320KB WGP LDS)
  (void)hipFuncSetAttribute((const void*)lstm_layer<640,128>,
                            hipFuncAttributeMaxDynamicSharedMemorySize, (int)sh0);
  (void)hipFuncSetAttribute((const void*)lstm_layer<1536,1024>,
                            hipFuncAttributeMaxDynamicSharedMemorySize, (int)sh1);
  lstm_layer<640,128>  <<<2, 1024, sh0, stream>>>(Wp0, bnx, g_g0, g_b0, c_g0, c_b0, h0);
  lstm_layer<1536,1024><<<2, 1024, sh1, stream>>>(Wp1, h0, g_g1, g_b1, c_g1, c_b1, h1);
  // BN2 + attention
  bn2_stats<<<2*Hh, 256, 0, stream>>>(h1, mu2, istd2);
  attn_out<<<Bb, 256, 0, stream>>>(h1, mu2, istd2, bn2_g, bn2_b, w_om, b_om, u_om, out);
}
